// VDEncoder_78889959292936
// MI455X (gfx1250) — 16-bit and 8-bit matrix operand order repaired and checked by reading, compiled, not run on hardware
//
#include <hip/hip_runtime.h>
#include <hip/hip_bf16.h>
#include <stdint.h>

#define BB   64
#define TT   2048
#define II   128
#define HH   256
#define G4H  1024
#define BT   (BB*TT)

typedef __attribute__((ext_vector_type(16))) __bf16       bf16x16;
typedef __attribute__((ext_vector_type(8)))  float        f32x8;
typedef __attribute__((ext_vector_type(8)))  int          i32x8;
typedef __attribute__((ext_vector_type(4)))  unsigned int u32x4;
typedef __attribute__((ext_vector_type(2)))  unsigned int u32x2;
typedef __attribute__((ext_vector_type(4)))  float        f32x4;

union FragBF { bf16x16 v; u32x4 q[2]; };
union Frag8  { i32x8  v; u32x4 q[2]; u32x2 d2[4]; };

__device__ __forceinline__ unsigned pack2(float a, float b) {
  union { __bf16 h[2]; unsigned u; } p;
  p.h[0] = (__bf16)a; p.h[1] = (__bf16)b;
  return p.u;
}
__device__ __forceinline__ float sigm(float x) {
  return 1.f / (1.f + __expf(-x));
}
__device__ __forceinline__ float tanh_f(float x) {
  x = fminf(fmaxf(x, -15.f), 15.f);
  float e = __expf(2.f * x);
  return (e - 1.f) / (e + 1.f);
}

__device__ __forceinline__ unsigned char f32_to_fp8(float x) {
#if __has_builtin(__builtin_amdgcn_cvt_pk_fp8_f32)
  return (unsigned char)(__builtin_amdgcn_cvt_pk_fp8_f32(x, x, 0, false) & 0xff);
#else
  unsigned u = __float_as_uint(x);
  unsigned s = (u >> 24) & 0x80u;
  unsigned au = u & 0x7fffffffu;
  if (au > 0x43e00000u) au = 0x43e00000u;
  if (au < 0x3c800000u) return (unsigned char)s;
  unsigned lsb = (au >> 20) & 1u;
  au += 0x0007ffffu + lsb;
  unsigned e = (au >> 23) - 127u + 7u;
  unsigned m = (au >> 20) & 7u;
  unsigned r = (e << 3) | m;
  if (r > 0x7eu) r = 0x7eu;
  return (unsigned char)(s | r);
#endif
}

__global__ void k_bias(const float* __restrict__ bi, const float* __restrict__ bh,
                       float* __restrict__ out) {
  int i = threadIdx.x + blockIdx.x * blockDim.x;
  if (i < G4H) out[i] = bi[i] + bh[i];
}

__global__ void k_prep_wih(const float* __restrict__ W, unsigned* __restrict__ out, int K) {
  int id = threadIdx.x + blockIdx.x * blockDim.x;
  int KC = K >> 5;
  if (id >= 64 * KC * 256) return;
  int v = id & 7, L = (id >> 3) & 31, f = id >> 8;
  int nt = f / KC, kc = f - nt * KC;
  int col = nt * 16 + (L & 15);
  int k   = kc * 32 + ((v >> 2) << 4) + ((L >> 4) << 3) + (v & 3) * 2;
  out[id] = pack2(W[col * K + k], W[col * K + k + 1]);
}

__global__ void k_prep_whh8(const float* __restrict__ W, unsigned* __restrict__ out) {
  int id = threadIdx.x + blockIdx.x * blockDim.x;
  int v = id & 7, L = (id >> 3) & 31, f = id >> 8;
  int tile = f >> 2, kc = f & 3;
  int ng = tile >> 3, rem = tile & 7, g4 = rem >> 1, u = rem & 1;
  int col = g4 * 256 + ng * 32 + u * 16 + (L & 15);
  int kb  = kc * 64 + ((v >> 1) << 4) + ((L >> 4) << 3) + ((v & 1) << 2);
  const float* src = W + col * HH + kb;
  unsigned d = (unsigned)f32_to_fp8(src[0])
             | ((unsigned)f32_to_fp8(src[1]) << 8)
             | ((unsigned)f32_to_fp8(src[2]) << 16)
             | ((unsigned)f32_to_fp8(src[3]) << 24);
  out[id] = d;
}

__device__ __forceinline__ long xw_index(long row  , int col) {
  int m = (int)(row >> 11), t = (int)(row & 2047);
  int mt = m >> 4, mm = m & 15, half = mm >> 3, r = mm & 7;
  int g4 = col >> 8, jj = col & 255;
  int ng = jj >> 5, u = (jj >> 4) & 1, nc = jj & 15;
  int L = half * 16 + nc;
  int w = ng * 4 + mt;
  int tl = g4 * 2 + u;
  return (((long)t * 32 + w) * 8 + tl) * 256 + L * 8 + r;
}

template <int KC, bool AF32>
__global__ __launch_bounds__(256) void k_gemm(const void* __restrict__ A_,
                                              const unsigned* __restrict__ Bs,
                                              const float* __restrict__ bias,
                                              float* __restrict__ C) {
  __shared__ __align__(16) __bf16 ash[64 * 40];
  const int K = KC * 32;
  const int tid = threadIdx.x, w = tid >> 5, L = tid & 31;
  const int mt = w & 3, ng = w >> 2;
  const long row0 = (long)blockIdx.x * 64;
  const int  nb0  = blockIdx.y * 16;

  f32x8 acc[8];
#pragma unroll
  for (int j = 0; j < 8; ++j)
#pragma unroll
    for (int r = 0; r < 8; ++r) acc[j][r] = 0.f;

  for (int kc = 0; kc < KC; ++kc) {
    {
      int r = tid >> 2, seg = tid & 3;
      u32x4* dst = (u32x4*)&ash[r * 40 + seg * 8];
      if (AF32) {
        const float* A = (const float*)A_ + (row0 + r) * K + kc * 32 + seg * 8;
        u32x4 d;
        d.x = pack2(A[0], A[1]); d.y = pack2(A[2], A[3]);
        d.z = pack2(A[4], A[5]); d.w = pack2(A[6], A[7]);
        *dst = d;
      } else {
        const __bf16* A = (const __bf16*)A_ + (row0 + r) * K + kc * 32 + seg * 8;
        *dst = *(const u32x4*)A;
      }
    }
    __syncthreads();
    FragBF a;
    const __bf16* ap = &ash[(mt * 16 + (L & 15)) * 40 + ((L >> 4) << 3)];
    a.q[0] = *(const u32x4*)ap;
    a.q[1] = *(const u32x4*)(ap + 16);
#pragma unroll
    for (int j = 0; j < 8; ++j) {
      int nt = nb0 + ng * 8 + j;
      const u32x4* bp = (const u32x4*)(Bs + (nt * KC + kc) * 256 + L * 8);
      FragBF b; b.q[0] = bp[0]; b.q[1] = bp[1];
      acc[j] = __builtin_amdgcn_wmma_f32_16x16x32_bf16(
          false, a.v, false, b.v, (short)0, acc[j], false, false);
    }
    __syncthreads();
  }
  const int  ncol  = L & 15;
  const long rbase = row0 + mt * 16 + ((L >> 4) << 3);
#pragma unroll
  for (int j = 0; j < 8; ++j) {
    int col = (nb0 + ng * 8 + j) * 16 + ncol;
    float bv = bias[col];
#pragma unroll
    for (int r = 0; r < 8; ++r)
      C[xw_index(rbase + r, col)] = acc[j][r] + bv;
  }
}

#define HPITCH 272
#define BSH_BYTES (256 * 1024)
__global__ __launch_bounds__(1024) void k_scan(const float* __restrict__ xW,
                                               const unsigned* __restrict__ Bs8,
                                               const float* __restrict__ mask,
                                               __bf16* __restrict__ outb,
                                               float*  __restrict__ outf,
                                               float* __restrict__ hn, float* __restrict__ cn) {
  extern __shared__ __align__(16) unsigned char smem[];
  unsigned char* Bsh = smem;
  unsigned char* hsh = smem + BSH_BYTES;

  const int tid = threadIdx.x, w = tid >> 5, L = tid & 31;
  const int mt = w & 3, ng = w >> 2;
  const int ncol = L & 15, mrow = (L >> 4) << 3;

  {
    const u32x4* src = (const u32x4*)Bs8;
    u32x4* dst = (u32x4*)Bsh;
    for (int i = tid; i < BSH_BYTES / 16; i += 1024) dst[i] = src[i];
    unsigned* hz = (unsigned*)hsh;
    for (int i = tid; i < 64 * HPITCH / 4; i += 1024) hz[i] = 0u;
  }

  float cc[2][8], mk[2][8];
#pragma unroll
  for (int u = 0; u < 2; ++u)
#pragma unroll
    for (int r = 0; r < 8; ++r) {
      cc[u][r] = 0.f;
      int m = mt * 16 + mrow + r;
      int j = ng * 32 + u * 16 + ncol;
      mk[u][r] = mask[m * HH + j];
    }
  __syncthreads();

  const unsigned char* habase = hsh + (mt * 16 + (L & 15)) * HPITCH + ((L >> 4) << 3);
  const unsigned char* bbase  = Bsh + (ng << 15) + L * 32;
  const long xwl = (long)w * 2048 + (long)L * 8;

  for (int t = 0; t < TT; ++t) {
    f32x8 acc[8];
#pragma unroll
    for (int tl = 0; tl < 8; ++tl) {
      const float* p = xW + ((long)t * 65536 + xwl + tl * 256);
      f32x4 lo = *(const f32x4*)p;
      f32x4 hi = *(const f32x4*)(p + 4);
      acc[tl][0] = lo.x; acc[tl][1] = lo.y; acc[tl][2] = lo.z; acc[tl][3] = lo.w;
      acc[tl][4] = hi.x; acc[tl][5] = hi.y; acc[tl][6] = hi.z; acc[tl][7] = hi.w;
    }
    if (t + 1 < TT) {
#pragma unroll
      for (int tl = 0; tl < 8; ++tl)
        __builtin_prefetch(xW + ((long)(t + 1) * 65536 + xwl + tl * 256), 0, 1);
    }
    for (int kc = 0; kc < 4; ++kc) {
      Frag8 a;
      const unsigned char* ap = habase + kc * 64;
#pragma unroll
      for (int g = 0; g < 4; ++g) a.d2[g] = *(const u32x2*)(ap + g * 16);

      Frag8 bg[4];
#pragma unroll
      for (int i = 0; i < 4; ++i) {
        const unsigned char* bb = bbase + ((i << 2) + kc) * 1024;
        bg[i].q[0] = *(const u32x4*)bb;
        bg[i].q[1] = *(const u32x4*)(bb + 16);
      }
#pragma unroll
      for (int i = 0; i < 4; ++i)
        acc[i] = __builtin_amdgcn_wmma_f32_16x16x64_fp8_fp8(
            a.v, bg[i].v, (short)0, acc[i], false, false);
#pragma unroll
      for (int i = 0; i < 4; ++i) {
        const unsigned char* bb = bbase + (((4 + i) << 2) + kc) * 1024;
        bg[i].q[0] = *(const u32x4*)bb;
        bg[i].q[1] = *(const u32x4*)(bb + 16);
      }
#pragma unroll
      for (int i = 0; i < 4; ++i)
        acc[4 + i] = __builtin_amdgcn_wmma_f32_16x16x64_fp8_fp8(
            a.v, bg[i].v, (short)0, acc[4 + i], false, false);
    }
    __syncthreads();
#pragma unroll
    for (int u = 0; u < 2; ++u)
#pragma unroll
      for (int r = 0; r < 8; ++r) {
        float ig = sigm(acc[0 + u][r]);
        float fg = sigm(acc[2 + u][r]);
        float gg = tanh_f(acc[4 + u][r]);
        float og = sigm(acc[6 + u][r]);
        float c  = fg * cc[u][r] + ig * gg;
        cc[u][r] = c;
        float h  = og * tanh_f(c);
        int m = mt * 16 + mrow + r;
        int j = ng * 32 + u * 16 + ncol;
        hsh[m * HPITCH + j] = f32_to_fp8(h);
        float ho = h * mk[u][r];
        long tok = (long)m * TT + t;
        if (outb) outb[tok * HH + j] = (__bf16)ho;
        else      outf[tok * HH + j] = ho;
        if (t == TT - 1) { hn[m * HH + j] = h; cn[m * HH + j] = c; }
      }
    __syncthreads();
  }
}

extern "C" void kernel_launch(void* const* d_in, const int* in_sizes, int n_in,
                              void* d_out, int out_size, void* d_ws, size_t ws_size,
                              hipStream_t stream) {
  (void)in_sizes; (void)n_in; (void)out_size; (void)ws_size;
  const float* x     = (const float*)d_in[0];
  const float* Wih0  = (const float*)d_in[1];
  const float* Whh0  = (const float*)d_in[2];
  const float* bih0  = (const float*)d_in[3];
  const float* bhh0  = (const float*)d_in[4];
  const float* mask0 = (const float*)d_in[5];
  const float* Wih1  = (const float*)d_in[6];
  const float* Whh1  = (const float*)d_in[7];
  const float* bih1  = (const float*)d_in[8];
  const float* bhh1  = (const float*)d_in[9];
  const float* mask1 = (const float*)d_in[10];

  char* ws = (char*)d_ws;
  size_t off = 0;
  auto carve = [&](size_t bytes) -> void* {
    void* p = ws + off;
    off = (off + bytes + 255) & ~(size_t)255;
    return p;
  };
  float*    xW    = (float*)carve((size_t)BT * G4H * 4);
  __bf16*   Xb1   = (__bf16*)carve((size_t)BT * HH * 2);
  unsigned* Wih0s = (unsigned*)carve((size_t)64 * 4 * 256 * 4);
  unsigned* Wih1s = (unsigned*)carve((size_t)64 * 8 * 256 * 4);
  unsigned* Whh0s = (unsigned*)carve((size_t)256 * 256 * 4);
  unsigned* Whh1s = (unsigned*)carve((size_t)256 * 256 * 4);
  float*    bias0 = (float*)carve(G4H * 4);
  float*    bias1 = (float*)carve(G4H * 4);

  float* out = (float*)d_out;
  float* hn0 = out + (size_t)BT * HH;
  float* hn1 = hn0 + (size_t)BB * HH;
  float* cn0 = hn1 + (size_t)BB * HH;
  float* cn1 = cn0 + (size_t)BB * HH;

  const size_t scan_lds = (size_t)BSH_BYTES + 64 * HPITCH;

  hipLaunchKernelGGL(k_bias, dim3(4), dim3(256), 0, stream, bih0, bhh0, bias0);
  hipLaunchKernelGGL(k_bias, dim3(4), dim3(256), 0, stream, bih1, bhh1, bias1);
  hipLaunchKernelGGL(k_prep_wih, dim3(64 * 4 * 256 / 256), dim3(256), 0, stream, Wih0, Wih0s, 128);
  hipLaunchKernelGGL(k_prep_wih, dim3(64 * 8 * 256 / 256), dim3(256), 0, stream, Wih1, Wih1s, 256);
  hipLaunchKernelGGL(k_prep_whh8, dim3(256), dim3(256), 0, stream, Whh0, Whh0s);
  hipLaunchKernelGGL(k_prep_whh8, dim3(256), dim3(256), 0, stream, Whh1, Whh1s);

  hipLaunchKernelGGL((k_gemm<4, true>),  dim3(2048, 4), dim3(256), 0, stream,
                     (const void*)x, Wih0s, bias0, xW);
  hipLaunchKernelGGL(k_scan, dim3(1), dim3(1024), scan_lds, stream,
                     xW, Whh0s, mask0, Xb1, (float*)nullptr, hn0, cn0);
  hipLaunchKernelGGL((k_gemm<8, false>), dim3(2048, 4), dim3(256), 0, stream,
                     (const void*)Xb1, Wih1s, bias1, xW);
  hipLaunchKernelGGL(k_scan, dim3(1), dim3(1024), scan_lds, stream,
                     xW, Whh1s, mask1, (__bf16*)nullptr, out, hn1, cn1);
}
